// MoMoShareLayer_9929964389225
// MI455X (gfx1250) — matrix operand order repaired and checked by reading, not compiled, not run on hardware
//
#include <hip/hip_runtime.h>
#include <math.h>

#define DEV __device__ __forceinline__

typedef __attribute__((ext_vector_type(16))) __bf16 v16bf;
typedef __attribute__((ext_vector_type(8)))  float  v8f;

constexpr int Bq = 16, Sq = 512, Dq = 768, Hq = 12, Fq = 3072;
constexpr int Eq = 4, Mq = 4, SWq = 128, DHq = 64;
constexpr int TOKq = Bq * Sq;

// ---------------- helpers ----------------

DEV unsigned pack2(float a, float b) {                   // two bf16 in one dword
  union { __bf16 h[2]; unsigned u; } c;
  c.h[0] = (__bf16)a; c.h[1] = (__bf16)b;
  return c.u;
}
DEV float geluf(float x) { return 0.5f * x * (1.0f + erff(x * 0.70710678118654752f)); }

DEV v8f wmma_bf16(v16bf a, v16bf b, v8f c) {
  return __builtin_amdgcn_wmma_f32_16x16x32_bf16(false, a, false, b, (short)0, c, false, false);
}

union FragU { v16bf v; uint4 q[2]; };

// A fragment: 16x32 bf16, LDS row-major [row][k], ld in elements (row stride bytes %16==0).
// lanes0-15: K=k0+0..7 (V0-3), K=k0+16..23 (V4-7); lanes16-31: +8.
DEV v16bf load_a_frag(const __bf16* p, int ld, int row0, int k0, int lane) {
  const __bf16* r = p + (row0 + (lane & 15)) * ld + k0 + ((lane >> 4) << 3);
  FragU u;
  u.q[0] = *(const uint4*)(r);        // 8 bf16
  u.q[1] = *(const uint4*)(r + 16);   // 8 bf16
  return u.v;
}
// B fragment: logical [K][N], stored transposed in LDS as [n][k] (ld elements per n-row).
// lanes0-15: K=k0+0..7, k0+16..23 for N=n0+lane; lanes16-31: +8.
DEV v16bf load_b_frag_t(const __bf16* p, int ld, int k0, int n0, int lane) {
  const __bf16* r = p + (n0 + (lane & 15)) * ld + k0 + ((lane >> 4) << 3);
  FragU u;
  u.q[0] = *(const uint4*)(r);
  u.q[1] = *(const uint4*)(r + 16);
  return u.v;
}

DEV void st8p(__bf16* d, float4 a, float4 b) {           // 8 floats -> 8 bf16, one 16B store
  uint4 u;
  u.x = pack2(a.x, a.y); u.y = pack2(a.z, a.w);
  u.z = pack2(b.x, b.y); u.w = pack2(b.z, b.w);
  *(uint4*)d = u;
}
DEV void st8z(__bf16* d) { uint4 u = {0,0,0,0}; *(uint4*)d = u; }

// Load a 64(k) x 64(n) fp32 weight tile, store TRANSPOSED bf16 into Bs[n][k] (ld=72).
DEV void load_w_t64(__bf16 (*Bs)[72], const float* __restrict__ Wz,
                    long long ldN, int k0, int n0, int tid) {
  const int n = tid & 63, kq = (tid >> 6) << 3;          // n 0..63, kq 0,8,16,24
#pragma unroll
  for (int half = 0; half < 2; ++half) {
    const int kk = kq + half * 32;
    const float* wp = Wz + (long long)(k0 + kk) * ldN + n0 + n;
    float v0 = wp[0];
    float v1 = wp[ldN];
    float v2 = wp[2*ldN];
    float v3 = wp[3*ldN];
    float v4 = wp[4*ldN];
    float v5 = wp[5*ldN];
    float v6 = wp[6*ldN];
    float v7 = wp[7*ldN];
    uint4 u;
    u.x = pack2(v0, v1); u.y = pack2(v2, v3); u.z = pack2(v4, v5); u.w = pack2(v6, v7);
    *(uint4*)&Bs[n][kk] = u;
  }
}

DEV float wave_sum(float v) {
#pragma unroll
  for (int o = 16; o > 0; o >>= 1) v += __shfl_xor(v, o, 32);
  return v;
}

// ---------------- generic 64x64-tile bf16-WMMA GEMM (K staged by 64) ----------------
// EPI: 0 = store, 1 = store gelu, 2 = C += scale[z]*(acc+bias)
template<int EPI>
__global__ __launch_bounds__(256) void gemm64(
    const float* __restrict__ A, const float* __restrict__ W,
    const float* __restrict__ bias, float* __restrict__ C,
    int rowsPerZ, int N, int K,
    const int* __restrict__ expIdx, long long wStride, long long bStride,
    const float* __restrict__ scale)
{
  __shared__ __align__(16) __bf16 As[64][72];
  __shared__ __align__(16) __bf16 Bs[64][72];   // transposed: [n][k]
  const int tid = threadIdx.x, lane = tid & 31, wv = tid >> 5;
  const int z = blockIdx.z, row0 = blockIdx.y * 64, n0 = blockIdx.x * 64;
  const float* Az = A + (long long)z * rowsPerZ * K;
  float* Cz = C + (long long)z * rowsPerZ * N;
  const int e = expIdx ? expIdx[z] : 0;
  const float* Wz = W + (long long)e * wStride;
  const float* bz = bias ? bias + (long long)e * bStride : nullptr;
  const int m_off = (wv >> 1) << 4, n_off = (wv & 1) << 5;
  v8f acc0 = {0,0,0,0,0,0,0,0}, acc1 = {0,0,0,0,0,0,0,0};
  const int lr = tid >> 2, lc = (tid & 3) << 4;   // A: 64 rows x 64 k, 16 floats/thread

  for (int k0 = 0; k0 < K; k0 += 64) {
    __syncthreads();
    {
      const float* ap = Az + (long long)(row0 + lr) * K + k0 + lc;
      st8p(&As[lr][lc],     *(const float4*)ap,       *(const float4*)(ap + 4));
      st8p(&As[lr][lc + 8], *(const float4*)(ap + 8), *(const float4*)(ap + 12));
      load_w_t64(Bs, Wz, N, k0, n0, tid);
    }
    __syncthreads();
#pragma unroll
    for (int kk = 0; kk < 64; kk += 32) {
      v16bf af = load_a_frag(&As[0][0], 72, m_off, kk, lane);
      v16bf b0 = load_b_frag_t(&Bs[0][0], 72, kk, n_off, lane);
      v16bf b1 = load_b_frag_t(&Bs[0][0], 72, kk, n_off + 16, lane);
      acc0 = wmma_bf16(af, b0, acc0);
      acc1 = wmma_bf16(af, b1, acc1);
    }
  }
  const float sc = (EPI == 2) ? (scale ? scale[z] : 1.0f) : 1.0f;
#pragma unroll
  for (int r = 0; r < 8; ++r) {
    const int Mg = row0 + m_off + r + ((lane >> 4) << 3);
    const int Ng = n0 + n_off + (lane & 15);
    float v0 = acc0[r] + (bz ? bz[Ng] : 0.0f);
    float v1 = acc1[r] + (bz ? bz[Ng + 16] : 0.0f);
    if (EPI == 1) { v0 = geluf(v0); v1 = geluf(v1); }
    float* c0 = Cz + (long long)Mg * N + Ng;
    if (EPI == 2) { c0[0] += sc * v0; c0[16] += sc * v1; }
    else          { c0[0]  = v0;      c0[16]  = v1; }
  }
}

// ---------------- flash-attention core (per b,h, 64-row q tile) ----------------
__global__ __launch_bounds__(256) void attn_core(
    const float* __restrict__ Q, const float* __restrict__ K,
    const float* __restrict__ V, const float* __restrict__ mask,
    float* __restrict__ CTX)
{
  const int qt = blockIdx.x, h = blockIdx.y, b = blockIdx.z;
  __shared__ __align__(16) __bf16 Qs[64][72];
  __shared__ __align__(16) __bf16 Ks[64][72];   // [key][dh] == B^T layout for QK^T
  __shared__ __align__(16) __bf16 Vts[64][40];  // transposed: [dh][key]
  __shared__ __align__(16) float  Sc[64][65];
  __shared__ __align__(16) __bf16 Ps[64][72];
  __shared__ float mrow[64], lrow[64], arow[64];
  const int tid = threadIdx.x, lane = tid & 31, wv = tid >> 5;
  const int m_off = (wv >> 1) << 4, n_off = (wv & 1) << 5;
  const int lr = tid >> 2, lc = (tid & 3) << 4;   // 64 rows x 64 cols, 16 floats/thread
  {
    const float* qp = Q + (long long)(b * Sq + qt * 64 + lr) * Dq + h * DHq + lc;
#pragma unroll
    for (int i = 0; i < 16; i += 8)
      st8p(&Qs[lr][lc + i], *(const float4*)(qp + i), *(const float4*)(qp + i + 4));
  }
  if (tid < 64) { mrow[tid] = -3.0e38f; lrow[tid] = 0.0f; }
  v8f o0 = {0,0,0,0,0,0,0,0}, o1 = {0,0,0,0,0,0,0,0};
  const float* mb = mask + (long long)b * Sq;
  const int vn = tid & 63, vkq = (tid >> 6) << 3;  // for V^T staging

  for (int kt = 0; kt < Sq; kt += 64) {
    __syncthreads();
    {
      const float* kp = K + (long long)(b * Sq + kt + lr) * Dq + h * DHq + lc;
#pragma unroll
      for (int i = 0; i < 16; i += 8)
        st8p(&Ks[lr][lc + i], *(const float4*)(kp + i), *(const float4*)(kp + i + 4));
      const float* vp = V + (long long)(b * Sq + kt + vkq) * Dq + h * DHq + vn;
      uint4 u;
      u.x = pack2(vp[0],      vp[Dq]);
      u.y = pack2(vp[2 * Dq], vp[3 * Dq]);
      u.z = pack2(vp[4 * Dq], vp[5 * Dq]);
      u.w = pack2(vp[6 * Dq], vp[7 * Dq]);
      *(uint4*)&Vts[vn][vkq] = u;
    }
    __syncthreads();
    // S = Q @ K^T
    v8f s0 = {0,0,0,0,0,0,0,0}, s1 = {0,0,0,0,0,0,0,0};
#pragma unroll
    for (int k0 = 0; k0 < DHq; k0 += 32) {
      v16bf qa  = load_a_frag(&Qs[0][0], 72, m_off, k0, lane);
      v16bf kb0 = load_b_frag_t(&Ks[0][0], 72, k0, n_off, lane);
      v16bf kb1 = load_b_frag_t(&Ks[0][0], 72, k0, n_off + 16, lane);
      s0 = wmma_bf16(qa, kb0, s0);
      s1 = wmma_bf16(qa, kb1, s1);
    }
#pragma unroll
    for (int r = 0; r < 8; ++r) {
      const int Mg = m_off + r + ((lane >> 4) << 3);
      Sc[Mg][n_off + (lane & 15)]      = s0[r];
      Sc[Mg][n_off + 16 + (lane & 15)] = s1[r];
    }
    __syncthreads();
    if (tid < 64) {       // online softmax, one row per thread
      float mold = mrow[tid], mx = mold;
      for (int c = 0; c < 64; ++c)
        mx = fmaxf(mx, Sc[tid][c] * 0.125f + mb[kt + c]);
      float alpha = __expf(mold - mx);
      float l = lrow[tid] * alpha;
      for (int c = 0; c < 64; c += 2) {
        float p0 = __expf(Sc[tid][c]     * 0.125f + mb[kt + c]     - mx);
        float p1 = __expf(Sc[tid][c + 1] * 0.125f + mb[kt + c + 1] - mx);
        *(unsigned*)&Ps[tid][c] = pack2(p0, p1);
        l += p0 + p1;
      }
      mrow[tid] = mx; lrow[tid] = l; arow[tid] = alpha;
    }
    __syncthreads();
#pragma unroll
    for (int r = 0; r < 8; ++r) {
      const float al = arow[m_off + r + ((lane >> 4) << 3)];
      o0[r] *= al; o1[r] *= al;
    }
    // O += P @ V
#pragma unroll
    for (int k0 = 0; k0 < 64; k0 += 32) {
      v16bf pa  = load_a_frag(&Ps[0][0], 72, m_off, k0, lane);
      v16bf vb0 = load_b_frag_t(&Vts[0][0], 40, k0, n_off, lane);
      v16bf vb1 = load_b_frag_t(&Vts[0][0], 40, k0, n_off + 16, lane);
      o0 = wmma_bf16(pa, vb0, o0);
      o1 = wmma_bf16(pa, vb1, o1);
    }
  }
#pragma unroll
  for (int r = 0; r < 8; ++r) {
    const int Mg = m_off + r + ((lane >> 4) << 3);
    const float inv = 1.0f / lrow[Mg];
    float* cp = CTX + (long long)(b * Sq + qt * 64 + Mg) * Dq + h * DHq + n_off + (lane & 15);
    cp[0]  = o0[r] * inv;
    cp[16] = o1[r] * inv;
  }
}

// ---------------- sequence-level router ----------------
__global__ __launch_bounds__(256) void router_kernel(
    const float* __restrict__ hs, const float* __restrict__ ew, const float* __restrict__ eb,
    const float* __restrict__ sw, const float* __restrict__ sb,
    int* __restrict__ selIdx, float* __restrict__ pm)
{
  const int b = blockIdx.x, tid = threadIdx.x;
  __shared__ float pooled[Dq];
  __shared__ float h1[SWq];
  __shared__ float lg[Eq];
  for (int c = tid; c < Dq; c += 256) {
    float s = 0.0f;
    const float* p = hs + (long long)b * Sq * Dq + c;
    for (int t = 0; t < Sq; ++t) s += p[(long long)t * Dq];
    pooled[c] = s * (1.0f / Sq);
  }
  __syncthreads();
  for (int o = tid; o < SWq; o += 256) {
    float s = eb[o];
    for (int k = 0; k < Dq; ++k) s += pooled[k] * ew[k * SWq + o];
    h1[o] = s;
  }
  __syncthreads();
  if (tid < Eq) {
    float s = sb[tid];
    for (int k = 0; k < SWq; ++k) s += h1[k] * sw[k * Eq + tid];
    lg[tid] = s;
  }
  __syncthreads();
  if (tid == 0) {
    float mx = lg[0]; int am = 0;
    for (int i = 1; i < Eq; ++i) if (lg[i] > mx) { mx = lg[i]; am = i; }
    float sum = 0.0f;
    for (int i = 0; i < Eq; ++i) sum += __expf(lg[i] - mx);
    selIdx[b] = am;
    pm[b] = 1.0f / sum;          // max softmax prob
  }
}

// ---------------- per-token switch router (one wave per token) ----------------
__global__ __launch_bounds__(256) void tok_router(
    const float* __restrict__ X, const float* __restrict__ uw, const float* __restrict__ ub,
    const int* __restrict__ selIdx, int* __restrict__ mIdx, float* __restrict__ gate)
{
  const int tok = blockIdx.x * 8 + (threadIdx.x >> 5);
  const int lane = threadIdx.x & 31;
  const int b = tok >> 9;
  const int e = selIdx[b];
  const float* x = X + (long long)tok * Dq;
  const float* w = uw + (long long)e * Dq * Mq;
  float a0 = 0, a1 = 0, a2 = 0, a3 = 0;
  for (int k = lane; k < Dq; k += 32) {
    const float xv = x[k];
    const float* wr = w + k * Mq;
    a0 += xv * wr[0]; a1 += xv * wr[1]; a2 += xv * wr[2]; a3 += xv * wr[3];
  }
  a0 = wave_sum(a0); a1 = wave_sum(a1); a2 = wave_sum(a2); a3 = wave_sum(a3);
  if (lane == 0) {
    float l[4] = { a0 + ub[e*Mq+0], a1 + ub[e*Mq+1], a2 + ub[e*Mq+2], a3 + ub[e*Mq+3] };
    float mx = l[0]; int am = 0;
    for (int i = 1; i < 4; ++i) if (l[i] > mx) { mx = l[i]; am = i; }
    float sum = 0.0f;
    for (int i = 0; i < 4; ++i) sum += __expf(l[i] - mx);
    mIdx[tok] = am;
    gate[tok] = 1.0f / sum;
  }
}

// ---------------- build per-(b,m) token lists ----------------
__global__ __launch_bounds__(256) void build_lists(
    const int* __restrict__ mIdx, int* __restrict__ counts,
    int* __restrict__ baseo, int* __restrict__ lists)
{
  const int b = blockIdx.x, tid = threadIdx.x;
  __shared__ int cnt[Mq];
  if (tid < Mq) cnt[tid] = 0;
  __syncthreads();
  for (int t = tid; t < Sq; t += 256) {
    const int m = mIdx[b * Sq + t];
    const int slot = atomicAdd(&cnt[m], 1);
    lists[(b * Mq + m) * Sq + slot] = t;
  }
  __syncthreads();
  if (tid == 0) {
    int base = 0;
    for (int m = 0; m < Mq; ++m) { counts[b*Mq+m] = cnt[m]; baseo[b*Mq+m] = base; base += cnt[m]; }
  }
}

// ---------------- routed MoE up: H[packed] = gelu(gather(X) @ W1[e,m] + b1) ----------------
__global__ __launch_bounds__(256) void moe_up(
    const float* __restrict__ X, const float* __restrict__ W1, const float* __restrict__ B1,
    const int* __restrict__ selIdx, const int* __restrict__ counts,
    const int* __restrict__ baseo, const int* __restrict__ lists,
    float* __restrict__ H)
{
  const int bm = blockIdx.z, b = bm >> 2, m = bm & 3;
  const int cnt = counts[bm];
  const int row0 = blockIdx.y * 64;
  if (row0 >= cnt) return;
  const int n0 = blockIdx.x * 64;
  const int e = selIdx[b];
  const int* lst = lists + bm * Sq;
  const int hbase = b * Sq + baseo[bm];
  const float* Wz = W1 + ((long long)(e * Mq + m) * Dq) * Fq;
  const float* bz = B1 + (long long)(e * Mq + m) * Fq;
  __shared__ __align__(16) __bf16 As[64][72];
  __shared__ __align__(16) __bf16 Bs[64][72];
  const int tid = threadIdx.x, lane = tid & 31, wv = tid >> 5;
  const int m_off = (wv >> 1) << 4, n_off = (wv & 1) << 5;
  v8f acc0 = {0,0,0,0,0,0,0,0}, acc1 = {0,0,0,0,0,0,0,0};
  const int lr = tid >> 2, lc = (tid & 3) << 4;
  const int rg = row0 + lr;
  const float* xrow = (rg < cnt) ? X + ((long long)b * Sq + lst[rg]) * Dq : nullptr;

  for (int k0 = 0; k0 < Dq; k0 += 64) {
    __syncthreads();
    if (xrow) {
      const float* ap = xrow + k0 + lc;
      st8p(&As[lr][lc],     *(const float4*)ap,       *(const float4*)(ap + 4));
      st8p(&As[lr][lc + 8], *(const float4*)(ap + 8), *(const float4*)(ap + 12));
    } else {
      st8z(&As[lr][lc]); st8z(&As[lr][lc + 8]);
    }
    load_w_t64(Bs, Wz, Fq, k0, n0, tid);
    __syncthreads();
#pragma unroll
    for (int kk = 0; kk < 64; kk += 32) {
      v16bf af = load_a_frag(&As[0][0], 72, m_off, kk, lane);
      v16bf b0 = load_b_frag_t(&Bs[0][0], 72, kk, n_off, lane);
      v16bf b1 = load_b_frag_t(&Bs[0][0], 72, kk, n_off + 16, lane);
      acc0 = wmma_bf16(af, b0, acc0);
      acc1 = wmma_bf16(af, b1, acc1);
    }
  }
#pragma unroll
  for (int r = 0; r < 8; ++r) {
    const int Ml = m_off + r + ((lane >> 4) << 3);
    const int rgE = row0 + Ml;
    if (rgE < cnt) {
      const int Ng = n0 + n_off + (lane & 15);
      float* hrow = H + (long long)(hbase + rgE) * Fq;
      hrow[Ng]      = geluf(acc0[r] + bz[Ng]);
      hrow[Ng + 16] = geluf(acc1[r] + bz[Ng + 16]);
    }
  }
}

// ---------------- routed MoE down: FFN[token] += gate * (H[packed] @ W2[e,m] + b2) ----------------
__global__ __launch_bounds__(256) void moe_down(
    const float* __restrict__ H, const float* __restrict__ W2, const float* __restrict__ B2,
    const int* __restrict__ selIdx, const int* __restrict__ counts,
    const int* __restrict__ baseo, const int* __restrict__ lists,
    const float* __restrict__ gate, float* __restrict__ FFN)
{
  const int bm = blockIdx.z, b = bm >> 2, m = bm & 3;
  const int cnt = counts[bm];
  const int row0 = blockIdx.y * 64;
  if (row0 >= cnt) return;
  const int n0 = blockIdx.x * 64;
  const int e = selIdx[b];
  const int* lst = lists + bm * Sq;
  const int hbase = b * Sq + baseo[bm];
  const float* Wz = W2 + ((long long)(e * Mq + m) * Fq) * Dq;
  const float* bz = B2 + (long long)(e * Mq + m) * Dq;
  __shared__ __align__(16) __bf16 As[64][72];
  __shared__ __align__(16) __bf16 Bs[64][72];
  const int tid = threadIdx.x, lane = tid & 31, wv = tid >> 5;
  const int m_off = (wv >> 1) << 4, n_off = (wv & 1) << 5;
  v8f acc0 = {0,0,0,0,0,0,0,0}, acc1 = {0,0,0,0,0,0,0,0};
  const int lr = tid >> 2, lc = (tid & 3) << 4;
  const int rg = row0 + lr;
  const float* hrow = (rg < cnt) ? H + (long long)(hbase + rg) * Fq : nullptr;

  for (int k0 = 0; k0 < Fq; k0 += 64) {
    __syncthreads();
    if (hrow) {
      const float* ap = hrow + k0 + lc;
      st8p(&As[lr][lc],     *(const float4*)ap,       *(const float4*)(ap + 4));
      st8p(&As[lr][lc + 8], *(const float4*)(ap + 8), *(const float4*)(ap + 12));
    } else {
      st8z(&As[lr][lc]); st8z(&As[lr][lc + 8]);
    }
    load_w_t64(Bs, Wz, Dq, k0, n0, tid);
    __syncthreads();
#pragma unroll
    for (int kk = 0; kk < 64; kk += 32) {
      v16bf af = load_a_frag(&As[0][0], 72, m_off, kk, lane);
      v16bf b0 = load_b_frag_t(&Bs[0][0], 72, kk, n_off, lane);
      v16bf b1 = load_b_frag_t(&Bs[0][0], 72, kk, n_off + 16, lane);
      acc0 = wmma_bf16(af, b0, acc0);
      acc1 = wmma_bf16(af, b1, acc1);
    }
  }
#pragma unroll
  for (int r = 0; r < 8; ++r) {
    const int Ml = m_off + r + ((lane >> 4) << 3);
    const int rgE = row0 + Ml;
    if (rgE < cnt) {
      const int tok = lst[rgE];
      const float g = gate[b * Sq + tok];
      const int Ng = n0 + n_off + (lane & 15);
      float* orow = FFN + ((long long)b * Sq + tok) * Dq;
      orow[Ng]      += g * (acc0[r] + bz[Ng]);
      orow[Ng + 16] += g * (acc1[r] + bz[Ng + 16]);
    }
  }
}

// ---------------- fused residual add + LayerNorm ----------------
__global__ __launch_bounds__(256) void ln_kernel(
    const float* __restrict__ A, const float* __restrict__ Fb,
    const float* __restrict__ g, const float* __restrict__ be, float* __restrict__ out)
{
  const int row = blockIdx.x, tid = threadIdx.x, lane = tid & 31, wv = tid >> 5;
  __shared__ float red[8];
  __shared__ float s_mu, s_rs;
  const float* a = A  + (long long)row * Dq;
  const float* f = Fb + (long long)row * Dq;
  float x0 = a[tid] + f[tid];
  float x1 = a[tid + 256] + f[tid + 256];
  float x2 = a[tid + 512] + f[tid + 512];
  float s = wave_sum(x0 + x1 + x2);
  if (lane == 0) red[wv] = s;
  __syncthreads();
  if (tid == 0) {
    float t = 0; for (int i = 0; i < 8; ++i) t += red[i];
    s_mu = t * (1.0f / Dq);
  }
  __syncthreads();
  const float mu = s_mu;
  const float d0 = x0 - mu, d1 = x1 - mu, d2 = x2 - mu;
  float s2 = wave_sum(d0*d0 + d1*d1 + d2*d2);
  if (lane == 0) red[wv] = s2;
  __syncthreads();
  if (tid == 0) {
    float t = 0; for (int i = 0; i < 8; ++i) t += red[i];
    s_rs = rsqrtf(t * (1.0f / Dq) + 1e-12f);
  }
  __syncthreads();
  const float rs = s_rs;
  float* o = out + (long long)row * Dq;
  o[tid]       = d0 * rs * g[tid]       + be[tid];
  o[tid + 256] = d1 * rs * g[tid + 256] + be[tid + 256];
  o[tid + 512] = d2 * rs * g[tid + 512] + be[tid + 512];
}

// ---------------- host side ----------------
extern "C" void kernel_launch(void* const* d_in, const int* in_sizes, int n_in,
                              void* d_out, int out_size, void* d_ws, size_t ws_size,
                              hipStream_t stream) {
  (void)in_sizes; (void)n_in; (void)out_size; (void)ws_size;
  const float* hs       = (const float*)d_in[0];
  const float* amask    = (const float*)d_in[1];
  const float* sw_enc_w = (const float*)d_in[3];
  const float* sw_enc_b = (const float*)d_in[4];
  const float* sw_w     = (const float*)d_in[5];
  const float* sw_b     = (const float*)d_in[6];
  const float* ca_wq = (const float*)d_in[7],  *ca_bq = (const float*)d_in[8];
  const float* ca_wk = (const float*)d_in[9],  *ca_bk = (const float*)d_in[10];
  const float* ca_wv = (const float*)d_in[11], *ca_bv = (const float*)d_in[12];
  const float* ca_wo = (const float*)d_in[13], *ca_bo = (const float*)d_in[14];
  const float* ua_wq = (const float*)d_in[15], *ua_bq = (const float*)d_in[16];
  const float* ua_wk = (const float*)d_in[17], *ua_bk = (const float*)d_in[18];
  const float* ua_wv = (const float*)d_in[19], *ua_bv = (const float*)d_in[20];
  const float* ua_wo = (const float*)d_in[21], *ua_bo = (const float*)d_in[22];
  const float* cf_w1 = (const float*)d_in[23], *cf_b1 = (const float*)d_in[24];
  const float* cf_w2 = (const float*)d_in[25], *cf_b2 = (const float*)d_in[26];
  const float* uf_sw_w = (const float*)d_in[27], *uf_sw_b = (const float*)d_in[28];
  const float* uf_w1 = (const float*)d_in[29], *uf_b1 = (const float*)d_in[30];
  const float* uf_w2 = (const float*)d_in[31], *uf_b2 = (const float*)d_in[32];
  const float* ln_g  = (const float*)d_in[33], *ln_b  = (const float*)d_in[34];
  float* out = (float*)d_out;

  // workspace carve-up (256B aligned)
  char* w = (char*)d_ws;
  size_t off = 0;
  auto take = [&](size_t bytes) -> void* {
    void* r = w + off;
    off = (off + bytes + 255) & ~(size_t)255;
    return r;
  };
  int*   selIdx = (int*)  take(Bq * 4);
  float* pm     = (float*)take(Bq * 4);
  int*   mIdx   = (int*)  take((size_t)TOKq * 4);
  float* gate   = (float*)take((size_t)TOKq * 4);
  int*   counts = (int*)  take(Bq * Mq * 4);
  int*   baseo  = (int*)  take(Bq * Mq * 4);
  int*   lists  = (int*)  take((size_t)Bq * Mq * Sq * 4);
  const size_t TD = (size_t)TOKq * Dq * 4;
  float* Qb  = (float*)take(TD);
  float* Kb  = (float*)take(TD);
  float* Vb  = (float*)take(TD);
  float* CTX = (float*)take(TD);
  float* ATT = (float*)take(TD);
  float* FFN = (float*)take(TD);
  float* Hb  = (float*)take((size_t)TOKq * Fq * 4);

  const long long DD = (long long)Dq * Dq;
  dim3 blk(256);
  dim3 gProj(Dq / 64, Sq / 64, Bq);     // 12 x 8 x 16
  dim3 gAttn(Sq / 64, Hq, Bq);          //  8 x 12 x 16
  dim3 gUp(Fq / 64, Sq / 64, Bq);       // 48 x 8 x 16
  dim3 gMoeUp(Fq / 64, Sq / 64, Bq * Mq);
  dim3 gMoeDn(Dq / 64, Sq / 64, Bq * Mq);

  // 1) sequence router
  router_kernel<<<Bq, blk, 0, stream>>>(hs, sw_enc_w, sw_enc_b, sw_w, sw_b, selIdx, pm);
  // 2) common attention
  gemm64<0><<<gProj, blk, 0, stream>>>(hs, ca_wq, ca_bq, Qb, Sq, Dq, Dq, nullptr, 0, 0, nullptr);
  gemm64<0><<<gProj, blk, 0, stream>>>(hs, ca_wk, ca_bk, Kb, Sq, Dq, Dq, nullptr, 0, 0, nullptr);
  gemm64<0><<<gProj, blk, 0, stream>>>(hs, ca_wv, ca_bv, Vb, Sq, Dq, Dq, nullptr, 0, 0, nullptr);
  attn_core<<<gAttn, blk, 0, stream>>>(Qb, Kb, Vb, amask, CTX);
  gemm64<0><<<gProj, blk, 0, stream>>>(CTX, ca_wo, ca_bo, ATT, Sq, Dq, Dq, nullptr, 0, 0, nullptr);
  // 3) selected unique attention, scaled by pm, added into ATT
  gemm64<0><<<gProj, blk, 0, stream>>>(hs, ua_wq, ua_bq, Qb, Sq, Dq, Dq, selIdx, DD, Dq, nullptr);
  gemm64<0><<<gProj, blk, 0, stream>>>(hs, ua_wk, ua_bk, Kb, Sq, Dq, Dq, selIdx, DD, Dq, nullptr);
  gemm64<0><<<gProj, blk, 0, stream>>>(hs, ua_wv, ua_bv, Vb, Sq, Dq, Dq, selIdx, DD, Dq, nullptr);
  attn_core<<<gAttn, blk, 0, stream>>>(Qb, Kb, Vb, amask, CTX);
  gemm64<2><<<gProj, blk, 0, stream>>>(CTX, ua_wo, ua_bo, ATT, Sq, Dq, Dq, selIdx, DD, Dq, pm);
  // 4) common FFN
  gemm64<1><<<gUp,   blk, 0, stream>>>(ATT, cf_w1, cf_b1, Hb,  Sq, Fq, Dq, nullptr, 0, 0, nullptr);
  gemm64<0><<<gProj, blk, 0, stream>>>(Hb,  cf_w2, cf_b2, FFN, Sq, Dq, Fq, nullptr, 0, 0, nullptr);
  // 5) unique switch-FFN (token-routed, top-1 of M inner experts)
  tok_router<<<TOKq / 8, blk, 0, stream>>>(ATT, uf_sw_w, uf_sw_b, selIdx, mIdx, gate);
  build_lists<<<Bq, blk, 0, stream>>>(mIdx, counts, baseo, lists);
  moe_up<<<gMoeUp, blk, 0, stream>>>(ATT, uf_w1, uf_b1, selIdx, counts, baseo, lists, Hb);
  moe_down<<<gMoeDn, blk, 0, stream>>>(Hb, uf_w2, uf_b2, selIdx, counts, baseo, lists, gate, FFN);
  // 6) out = LayerNorm(ATT + FFN)
  ln_kernel<<<TOKq, blk, 0, stream>>>(ATT, FFN, ln_g, ln_b, out);
}
